// VGRUatt_28346784153998
// MI455X (gfx1250) — matrix operand order repaired and checked by reading, not compiled, not run on hardware
//
#include <hip/hip_runtime.h>
#include <hip/hip_bf16.h>
#include <cstdint>

// ---------------------------------------------------------------------------
// Fused GRU-attention decoder for MI455X (gfx1250, wave32, WMMA 16x16x32 f16).
// B=8192, T=25, D=128, V=78.
//   Kernel 0: convert weights + seq fp32 -> f16 into d_ws.
//   Kernel 1: persistent recurrence, 16 batch rows / block, 8 waves.
//             Block's seq slice (100KB f16) DMA'd into LDS once via the
//             Tensor Data Mover (tensor_load_to_lds + s_wait_tensorcnt);
//             all per-step GEMMs via v_wmma_f32_16x16x32_f16; gate math in
//             registers; attention phases read seq from LDS (no global
//             re-reads across the 25 steps).
//   Kernel 2: outs @ W_out + row softmax (V=78), 4 WMMAs per 16x16 tile.
// ---------------------------------------------------------------------------

typedef _Float16 v16h __attribute__((ext_vector_type(16)));
typedef float    v8f  __attribute__((ext_vector_type(8)));
typedef unsigned int u32x4 __attribute__((ext_vector_type(4)));
typedef int          i32x8 __attribute__((ext_vector_type(8)));
typedef int          i32x4 __attribute__((ext_vector_type(4)));

static constexpr int Bn = 8192, Tn = 25, Dn = 128, Vn = 78;
static constexpr int MB = 16;          // batch rows per block
static constexpr int VP = 80;          // V padded to multiple of 16
static constexpr int SEQ_TILE = MB * Tn * Dn;   // 51200 f16 elems = 102400 B

__device__ __forceinline__ v8f wmma16(v16h a, v16h b, v8f c) {
  // (neg_a, A, neg_b, B, c_mod, C, reuse_a, reuse_b)
  return __builtin_amdgcn_wmma_f32_16x16x32_f16(false, a, false, b,
                                                (short)0, c, false, false);
}

// --- A fragment: 16x32 f16 tile, row-major source, leading dim ld (halfs) ---
// ISA 7.12.2: lane = half*16 + m ; VGPR v holds K = (v>=4?16:0)+half*8+(v%4)*2
__device__ __forceinline__ v16h load_a(const _Float16* base, int ld) {
  const int lane = threadIdx.x & 31;
  const int m = lane & 15, half = lane >> 4;
  const _Float16* row = base + m * ld;
  v16h a;
#pragma unroll
  for (int v = 0; v < 8; ++v) {
    const int kb = ((v & 4) << 2) + half * 8 + (v & 3) * 2;
    a[2 * v]     = row[kb];
    a[2 * v + 1] = row[kb + 1];
  }
  return a;
}

// --- B fragment: 32x16 f16 tile of row-major W[K][N] at (k0,n0), ld = N ----
// ISA 7.12.2 (B rows striped across lanes): lane = half*16 + n ;
// VGPR v holds K = (v>=4?16:0)+half*8+(v%4)*2 (+1)
__device__ __forceinline__ v16h load_b(const _Float16* w, int ld, int k0, int n0) {
  const int lane = threadIdx.x & 31;
  const int n = lane & 15, half = lane >> 4;
  v16h b;
#pragma unroll
  for (int v = 0; v < 8; ++v) {
    const int kb = ((v & 4) << 2) + (half << 3) + ((v & 3) << 1);
    b[2 * v]     = w[(k0 + kb) * ld + n0 + n];
    b[2 * v + 1] = w[(k0 + kb + 1) * ld + n0 + n];
  }
  return b;
}

// --- B fragment from transposed source: B[k][n] = W[n0+n][k0+k] ------------
__device__ __forceinline__ v16h load_bt(const _Float16* w, int ld, int k0, int n0) {
  const int lane = threadIdx.x & 31;
  const int n = lane & 15, half = lane >> 4;
  const _Float16* row = w + (n0 + n) * ld + k0;
  v16h b;
#pragma unroll
  for (int v = 0; v < 8; ++v) {
    const int kb = ((v & 4) << 2) + (half << 3) + ((v & 3) << 1);
    b[2 * v]     = row[kb];
    b[2 * v + 1] = row[kb + 1];
  }
  return b;
}

// ============================ Kernel 0: convert ============================
__global__ __launch_bounds__(256) void cvt_inputs(
    const float* __restrict__ w_att, const float* __restrict__ gk,
    const float* __restrict__ gr, const float* __restrict__ w_out,
    const float* __restrict__ seq, _Float16* __restrict__ w_att_h,
    _Float16* __restrict__ gk_h, _Float16* __restrict__ gr_h,
    _Float16* __restrict__ w_out_h, _Float16* __restrict__ seq_h) {
  const size_t i = (size_t)blockIdx.x * 256 + threadIdx.x;
  if (i < Dn * Dn) w_att_h[i] = (_Float16)w_att[i];
  if (i < Dn * 3 * Dn) {
    gk_h[i] = (_Float16)gk[i];
    gr_h[i] = (_Float16)gr[i];
  }
  if (i < Dn * VP) {
    const int r = (int)i / VP, c = (int)i % VP;
    w_out_h[i] = (c < Vn) ? (_Float16)w_out[r * Vn + c] : (_Float16)0.f;
  }
  const size_t n = (size_t)Bn * Tn * Dn;
  const size_t stride = (size_t)gridDim.x * 256;
  for (size_t j = i; j < n; j += stride) seq_h[j] = (_Float16)seq[j];
}

// ========================= Kernel 1: recurrence ============================
__global__ __launch_bounds__(256) void gru_att_recurrence(
    const _Float16* __restrict__ seq_h, const float* __restrict__ H,
    const float* __restrict__ x0, const _Float16* __restrict__ w_att_h,
    const float* __restrict__ b_att, const _Float16* __restrict__ gk_h,
    const _Float16* __restrict__ gr_h, const float* __restrict__ gbias,
    _Float16* __restrict__ outs) {
  __shared__ _Float16 s_seq[SEQ_TILE];     // block's seq slice, f16 (100KB)
  __shared__ _Float16 s_inp_h[MB * Dn];    // GRU input (f16, WMMA A operand)
  __shared__ _Float16 s_state_h[MB * Dn];  // GRU state (f16, WMMA A operand)
  __shared__ float    s_state[MB * Dn];    // GRU state (f32, gate blend)
  __shared__ float    s_out[MB * Dn];      // step output (f32)
  __shared__ _Float16 s_out_h[MB * Dn];    // step output (f16, A operand)
  __shared__ float    s_v[MB * Dn];        // v = out @ W_att^T
  __shared__ float    s_logit[MB * 32];    // logits -> alphas (T=25 pad 32)
  __shared__ float    s_c[MB];             // b_att . out

  const int tid  = threadIdx.x;
  const int wave = tid >> 5;
  const int lane = tid & 31;
  const long b0  = (long)blockIdx.x * MB;

  // ---- TDM: DMA this block's contiguous 102400B seq slice into LDS. ------
  // 1-D D#: data_size=2B, tensor_dim0 = tile_dim0 = 51200 elems, count=1.
  if (wave == 0) {
    const uint64_t ga =
        (uint64_t)(uintptr_t)(seq_h + (size_t)b0 * Tn * Dn);
    const uint32_t lds_off = (uint32_t)(uintptr_t)(&s_seq[0]);
    // group0: [1:0]=count=1 | [63:32]=lds_addr | [120:64]=global_addr |
    //         [127:126]=type=2
    u32x4 g0 = {1u, lds_off, (uint32_t)ga,
                (uint32_t)(ga >> 32) | 0x80000000u};
    // group1: w0: data_size=1(2B)<<16 ; w1[31:16]=tensor_dim0.lo16 ;
    //         w2[31:16]=tensor_dim1.lo16(=1) ; w3[31:16]=tile_dim0 ;
    //         w4[15:0]=tile_dim1(=1) ; w5=tensor_dim0_stride(=51200)
    i32x8 g1 = {0x00010000, (int)0xC8000000, 0x00010000, (int)0xC8000000,
                1, 51200, 0, 0};
    i32x4 gz = {0, 0, 0, 0};
#if __clang_major__ >= 23
    i32x8 gz8 = {0, 0, 0, 0, 0, 0, 0, 0};
    __builtin_amdgcn_tensor_load_to_lds(g0, g1, gz, gz, gz8, 0);
#else
    __builtin_amdgcn_tensor_load_to_lds(g0, g1, gz, gz, 0);
#endif
  }

  // init carry: inp = x0, state = H (overlaps with TDM transfer)
  for (int i = tid; i < MB * Dn; i += 256) {
    const int r = i >> 7, d = i & 127;
    const float xv = x0[(b0 + r) * Dn + d];
    const float hv = H[(b0 + r) * Dn + d];
    s_inp_h[i]   = (_Float16)xv;
    s_state[i]   = hv;
    s_state_h[i] = (_Float16)hv;
  }
  __builtin_amdgcn_s_wait_tensorcnt(0);
  __syncthreads();

  const int n0   = wave * 16;              // this wave's column tile in D
  const int coln = n0 + (lane & 15);       // column this lane owns in C tiles

  for (int t = 0; t < Tn; ++t) {
    // ---- Phase A: GRU gate GEMMs (weights live in VGPRs across steps) ----
    v8f accz = {}, accr = {}, accxh = {}, acchh = {};
#pragma unroll
    for (int k = 0; k < Dn; k += 32) {
      const v16h a_in = load_a(s_inp_h + k, Dn);
      const v16h a_st = load_a(s_state_h + k, Dn);
      accz  = wmma16(a_in, load_b(gk_h, 3 * Dn, k, n0), accz);
      accz  = wmma16(a_st, load_b(gr_h, 3 * Dn, k, n0), accz);
      accr  = wmma16(a_in, load_b(gk_h, 3 * Dn, k, Dn + n0), accr);
      accr  = wmma16(a_st, load_b(gr_h, 3 * Dn, k, Dn + n0), accr);
      accxh = wmma16(a_in, load_b(gk_h, 3 * Dn, k, 2 * Dn + n0), accxh);
      acchh = wmma16(a_st, load_b(gr_h, 3 * Dn, k, 2 * Dn + n0), acchh);
    }
    const float bz  = gbias[coln] + gbias[3 * Dn + coln];
    const float br  = gbias[Dn + coln] + gbias[3 * Dn + Dn + coln];
    const float bxh = gbias[2 * Dn + coln];
    const float bhh = gbias[3 * Dn + 2 * Dn + coln];

    if (tid < MB) s_c[tid] = 0.f;
#pragma unroll
    for (int i = 0; i < 8; ++i) {
      const int m  = i + ((lane >> 4) << 3);
      const float st   = s_state[m * Dn + coln];
      const float z    = 1.f / (1.f + __expf(-(accz[i] + bz)));
      const float r    = 1.f / (1.f + __expf(-(accr[i] + br)));
      const float cand = tanhf((accxh[i] + bxh) + r * (acchh[i] + bhh));
      const float ov   = z * st + (1.f - z) * cand;
      s_out[m * Dn + coln]   = ov;
      s_out_h[m * Dn + coln] = (_Float16)ov;
      outs[((size_t)(b0 + m) * Tn + t) * Dn + coln] = (_Float16)ov;
    }
    __syncthreads();

    // ---- Phase B: v = out @ W_att^T (per-wave 16x16 tile), c = b_att.out ---
    v8f accv = {};
#pragma unroll
    for (int k = 0; k < Dn; k += 32)
      accv = wmma16(load_a(s_out_h + k, Dn), load_bt(w_att_h, Dn, k, n0), accv);
#pragma unroll
    for (int i = 0; i < 8; ++i) {
      const int m = i + ((lane >> 4) << 3);
      s_v[m * Dn + coln] = accv[i];
    }
    {
      const int row = tid >> 4, chunk = tid & 15;
      float p = 0.f;
#pragma unroll
      for (int j = 0; j < 8; ++j) {
        const int d = chunk * 8 + j;
        p += b_att[d] * s_out[row * Dn + d];
      }
      atomicAdd(&s_c[row], p);
    }
    __syncthreads();

    // ---- Phase C: logits[b,t'] = seq[b,t',:] . v[b] + c[b]  (LDS only) ----
    for (int idx = tid; idx < MB * Tn; idx += 256) {
      const int row = idx / Tn, tt = idx % Tn;
      const _Float16* sp = s_seq + (row * Tn + tt) * Dn;
      const float* sv = s_v + row * Dn;
      float acc = 0.f;
#pragma unroll 8
      for (int j = 0; j < Dn; ++j) acc += (float)sp[j] * sv[j];
      s_logit[row * 32 + tt] = acc + s_c[row];
    }
    __syncthreads();

    // ---- Phase D: softmax over T=25 (one thread per batch row) ----
    if (tid < MB) {
      float mx = -3.4e38f;
      for (int tt = 0; tt < Tn; ++tt) mx = fmaxf(mx, s_logit[tid * 32 + tt]);
      float sum = 0.f;
      for (int tt = 0; tt < Tn; ++tt) {
        const float e = __expf(s_logit[tid * 32 + tt] - mx);
        s_logit[tid * 32 + tt] = e;
        sum += e;
      }
      const float inv = 1.f / sum;
      for (int tt = 0; tt < Tn; ++tt) s_logit[tid * 32 + tt] *= inv;
    }
    __syncthreads();

    // ---- Phase E: ctx = sum_t alpha*seq -> next state; next inp = out -----
    for (int i = tid; i < MB * Dn; i += 256) {
      const int row = i >> 7, d = i & 127;
      const _Float16* sq = s_seq + row * Tn * Dn + d;
      float acc = 0.f;
#pragma unroll
      for (int tt = 0; tt < Tn; ++tt)
        acc += s_logit[row * 32 + tt] * (float)sq[tt * Dn];
      s_state[i]   = acc;
      s_state_h[i] = (_Float16)acc;
      s_inp_h[i]   = s_out_h[i];
    }
    __syncthreads();
  }
}

// ==================== Kernel 2: projection + softmax =======================
__global__ __launch_bounds__(160) void out_proj_softmax(
    const _Float16* __restrict__ outs, const _Float16* __restrict__ w_out_h,
    const float* __restrict__ b_out, float* __restrict__ out) {
  __shared__ _Float16 s_a[MB * Dn];
  __shared__ float    s_log[MB * VP];

  const int tid  = threadIdx.x;
  const int wave = tid >> 5;           // 5 waves, one per 16-col tile of VP=80
  const int lane = tid & 31;
  const long row0 = (long)blockIdx.x * MB;   // rows of the (B*T) x D matrix

  for (int i = tid; i < MB * Dn; i += 160) s_a[i] = outs[(size_t)row0 * Dn + i];
  __syncthreads();

  const int n0 = wave * 16;
  v8f acc = {};
#pragma unroll
  for (int k = 0; k < Dn; k += 32)
    acc = wmma16(load_a(s_a + k, Dn), load_b(w_out_h, VP, k, n0), acc);
#pragma unroll
  for (int i = 0; i < 8; ++i) {
    const int m   = i + ((lane >> 4) << 3);
    const int col = n0 + (lane & 15);
    s_log[m * VP + col] = acc[i] + ((col < Vn) ? b_out[col] : 0.f);
  }
  __syncthreads();

  // two-pass softmax (recompute exp; no per-thread spill array)
  if (tid < MB) {
    const float* lr = s_log + tid * VP;
    float mx = -3.4e38f;
    for (int v = 0; v < Vn; ++v) mx = fmaxf(mx, lr[v]);
    float sum = 0.f;
    for (int v = 0; v < Vn; ++v) sum += __expf(lr[v] - mx);
    const float inv = 1.f / sum;
    float* orow = out + (size_t)(row0 + tid) * Vn;
    for (int v = 0; v < Vn; ++v) orow[v] = __expf(lr[v] - mx) * inv;
  }
}

// ================================ launch ===================================
extern "C" void kernel_launch(void* const* d_in, const int* in_sizes, int n_in,
                              void* d_out, int out_size, void* d_ws,
                              size_t ws_size, hipStream_t stream) {
  const float* seq   = (const float*)d_in[0];
  const float* H     = (const float*)d_in[1];
  const float* x0    = (const float*)d_in[2];
  const float* W_att = (const float*)d_in[3];
  const float* b_att = (const float*)d_in[4];
  const float* gk    = (const float*)d_in[5];
  const float* gr    = (const float*)d_in[6];
  const float* gb    = (const float*)d_in[7];
  const float* W_out = (const float*)d_in[8];
  const float* b_out = (const float*)d_in[9];

  char* ws = (char*)d_ws;
  _Float16* w_att_h = (_Float16*)(ws + 0);         //  32768 B
  _Float16* gk_h    = (_Float16*)(ws + 32768);     //  98304 B
  _Float16* gr_h    = (_Float16*)(ws + 131072);    //  98304 B
  _Float16* w_out_h = (_Float16*)(ws + 229376);    //  20480 B (128 x 80)
  _Float16* outs    = (_Float16*)(ws + 262144);    //  B*T*D f16 = 50 MB
  _Float16* seq_h   = (_Float16*)(ws + 262144 + (size_t)Bn * Tn * Dn * 2);

  // fp32 -> f16: weights + seq (grid-stride over 26.2M elems)
  cvt_inputs<<<4096, 256, 0, stream>>>(W_att, gk, gr, W_out, seq, w_att_h,
                                       gk_h, gr_h, w_out_h, seq_h);

  // fused recurrence: 8192/16 = 512 blocks, 8 waves each
  gru_att_recurrence<<<Bn / MB, 256, 0, stream>>>(seq_h, H, x0, w_att_h,
                                                  b_att, gk_h, gr_h, gb, outs);

  // projection + softmax: (B*T)/16 = 12800 blocks, 5 waves each
  out_proj_softmax<<<(Bn * Tn) / MB, 160, 0, stream>>>(outs, w_out_h, b_out,
                                                       (float*)d_out);
}
